// SelfAttentionModel_2276332667579
// MI455X (gfx1250) — compile-verified, matrix operand order checked by reading, not run on hardware
//
#include <hip/hip_runtime.h>
#include <hip/hip_bf16.h>

typedef __attribute__((ext_vector_type(16))) _Float16 v16h;
typedef __attribute__((ext_vector_type(8)))  _Float16 v8h;
typedef __attribute__((ext_vector_type(8)))  float    v8f;
typedef __attribute__((ext_vector_type(4)))  float    v4f;

#define SEQ 2048
#define EMBD 256

__device__ __forceinline__ v16h combine16(v8h lo, v8h hi) {
    return __builtin_shufflevector(lo, hi, 0,1,2,3,4,5,6,7,8,9,10,11,12,13,14,15);
}

__device__ __forceinline__ v8f wmma_f16(v16h a, v16h b, v8f c) {
    // D = A(16x32 f16) * B(32x16 f16) + C(16x16 f32)
    return __builtin_amdgcn_wmma_f32_16x16x32_f16(false, a, false, b, (short)0, c, false, false);
}

// ---------------------------------------------------------------------------
// Kernel 0: weight prep. Wt[n][k] = (f16) W[k][n]  for Q,K,V (3 x 256 x 256).
// B-fragments then read contiguous 16B runs in the K dimension.
// ---------------------------------------------------------------------------
__global__ __launch_bounds__(256) void prep_kernel(
    const float* __restrict__ Wq, const float* __restrict__ Wk,
    const float* __restrict__ Wv,
    _Float16* __restrict__ WtQ, _Float16* __restrict__ WtK,
    _Float16* __restrict__ WtV)
{
    int id = blockIdx.x * 256 + threadIdx.x;       // 0 .. 196607
    int mat = id >> 16;
    int rem = id & 65535;
    int n = rem >> 8, k = rem & 255;
    const float* W  = (mat == 0) ? Wq  : (mat == 1) ? Wk  : Wv;
    _Float16*    Wt = (mat == 0) ? WtQ : (mat == 1) ? WtK : WtV;
    Wt[n * 256 + k] = (_Float16)W[k * 256 + n];
}

// ---------------------------------------------------------------------------
// Kernel 1: embedding gather + Q/K/V projection (f16 WMMA, f32 accumulate).
// grid.x = (B*S)/16 blocks of 256 threads (8 waves); 48 tiles / 8 waves.
// Q,K stored row-major [token][h]; V stored transposed [b][h][s].
// ---------------------------------------------------------------------------
__global__ __launch_bounds__(256) void qkv_kernel(
    const int* __restrict__ x, const float* __restrict__ emb,
    const _Float16* __restrict__ WtQ, const _Float16* __restrict__ WtK,
    const _Float16* __restrict__ WtV,
    const float* __restrict__ bq, const float* __restrict__ bk,
    const float* __restrict__ bv,
    _Float16* __restrict__ Qh, _Float16* __restrict__ Kh,
    _Float16* __restrict__ Vt)
{
    __shared__ __align__(16) _Float16 sE[16][264];   // 16 token rows, f16
    const int tid = threadIdx.x;
    const long rowBase = (long)blockIdx.x * 16;

    // gather+convert: 16x256 f32 -> f16 LDS (coalesced across threads)
    #pragma unroll
    for (int i = 0; i < 16; ++i) {
        int idx = i * 256 + tid;
        int r = idx >> 8, c = idx & 255;
        int tok = x[rowBase + r];
        sE[r][c] = (_Float16)emb[(long)tok * EMBD + c];
    }
    __syncthreads();

    const int wave = tid >> 5, lane = tid & 31;
    const int lm = lane & 15, hh = lane >> 4;

    for (int j = 0; j < 6; ++j) {
        int tile = wave + 8 * j;                 // 0..47
        int mat  = tile >> 4;                    // 0=Q 1=K 2=V
        int n0   = (tile & 15) * 16;
        const _Float16* Wt   = (mat == 0) ? WtQ : (mat == 1) ? WtK : WtV;
        const float*    bias = (mat == 0) ? bq  : (mat == 1) ? bk  : bv;

        const _Float16* wp = Wt + (long)(n0 + lm) * EMBD;   // column n, k contiguous
        v8f acc = {};
        for (int kc = 0; kc < 8; ++kc) {
            int kb = kc * 32 + hh * 8;
            v16h A = combine16(*(const v8h*)&sE[lm][kb],
                               *(const v8h*)&sE[lm][kb + 16]);
            v16h B = combine16(*(const v8h*)&wp[kb],
                               *(const v8h*)&wp[kb + 16]);
            acc = wmma_f16(A, B, acc);
        }
        float bval = bias[n0 + lm];
        if (mat == 2) {
            // transposed store: Vt[b][h][s]; rows r map to consecutive s
            int bb = (int)(rowBase >> 11);
            int s0 = (int)(rowBase & (SEQ - 1)) + 8 * hh;
            v8h ov;
            #pragma unroll
            for (int r = 0; r < 8; ++r) ov[r] = (_Float16)(acc[r] + bval);
            *(v8h*)&Vt[((long)bb * EMBD + n0 + lm) * SEQ + s0] = ov;
        } else {
            _Float16* Out = (mat == 0) ? Qh : Kh;
            #pragma unroll
            for (int r = 0; r < 8; ++r) {
                int m = r + 8 * hh;
                Out[(rowBase + m) * EMBD + n0 + lm] = (_Float16)(acc[r] + bval);
            }
        }
    }
}

// ---------------------------------------------------------------------------
// Kernel 2: flash attention + fused mean-pool accumulation.
// grid = (128 query tiles, 32 batches), 256 threads (8 waves).
// Q A-fragments held in registers; K and V(transposed) read as b128 runs.
// ---------------------------------------------------------------------------
__global__ __launch_bounds__(256) void attn_kernel(
    const _Float16* __restrict__ Qh, const _Float16* __restrict__ Kh,
    const _Float16* __restrict__ Vt, float* __restrict__ pooled)
{
    __shared__ float sS[16][132];
    __shared__ __align__(16) _Float16 sP[16][136];
    __shared__ float rowM[16], rowL[16], rowScale[16];

    const int b = blockIdx.y, mt = blockIdx.x;
    const int tid = threadIdx.x;
    const int wave = tid >> 5, lane = tid & 31;
    const int lm = lane & 15, hh = lane >> 4;
    const int hcol = wave * 32;                 // this wave owns h cols [hcol, hcol+32)
    const long qRow = (long)b * SEQ + mt * 16;

    if (tid < 16) { rowM[tid] = -__builtin_inff(); rowL[tid] = 0.f; }

    // Q A-fragments: row = lm, K-dim contiguous in memory -> registers once.
    v16h qA[8];
    {
        const _Float16* qp = Qh + (qRow + lm) * EMBD;
        #pragma unroll
        for (int kc = 0; kc < 8; ++kc) {
            int kb = kc * 32 + hh * 8;
            qA[kc] = combine16(*(const v8h*)&qp[kb], *(const v8h*)&qp[kb + 16]);
        }
    }

    v8f acc0 = {}, acc1 = {};
    const _Float16* kpBase = Kh + ((long)b * SEQ + wave * 16 + lm) * EMBD;
    const _Float16* v0Base = Vt + ((long)b * EMBD + hcol + lm) * SEQ;
    const _Float16* v1Base = Vt + ((long)b * EMBD + hcol + 16 + lm) * SEQ;

    for (int ch = 0; ch < 16; ++ch) {
        // ---- scores: Q(16x256) @ K^T, this wave's 16 keys ----
        const _Float16* kp = kpBase + (long)ch * 128 * EMBD;
        if (ch < 15) {
            __builtin_prefetch((const void*)(kp + 128 * EMBD), 0, 0);
            __builtin_prefetch((const void*)(v0Base + (ch + 1) * 128), 0, 0);
            __builtin_prefetch((const void*)(v1Base + (ch + 1) * 128), 0, 0);
        }
        v8f sc = {};
        #pragma unroll
        for (int kc = 0; kc < 8; ++kc) {
            int kb = kc * 32 + hh * 8;
            v16h B = combine16(*(const v8h*)&kp[kb], *(const v8h*)&kp[kb + 16]);
            sc = wmma_f16(qA[kc], B, sc);
        }
        #pragma unroll
        for (int r = 0; r < 8; ++r)
            sS[r + 8 * hh][wave * 16 + lm] = sc[r] * 0.0625f;   // 1/sqrt(256)
        __syncthreads();

        // ---- online softmax over 128-key chunk (2 rows per wave) ----
        {
            int row = 2 * wave + hh;
            v4f sv0 = *(const v4f*)&sS[row][lm * 8];
            v4f sv1 = *(const v4f*)&sS[row][lm * 8 + 4];
            float v[8] = { sv0.x, sv0.y, sv0.z, sv0.w, sv1.x, sv1.y, sv1.z, sv1.w };
            float mloc = -__builtin_inff();
            #pragma unroll
            for (int jj = 0; jj < 8; ++jj) mloc = fmaxf(mloc, v[jj]);
            #pragma unroll
            for (int msk = 1; msk <= 8; msk <<= 1)
                mloc = fmaxf(mloc, __shfl_xor(mloc, msk, 32));
            float mold = rowM[row];
            float mnew = fmaxf(mold, mloc);
            float scl  = __expf(mold - mnew);     // exp(-inf)=0 on first chunk
            float lsum = 0.f;
            v8h pv;
            #pragma unroll
            for (int jj = 0; jj < 8; ++jj) {
                float p = __expf(v[jj] - mnew);
                pv[jj] = (_Float16)p;
                lsum += p;
            }
            *(v8h*)&sP[row][lm * 8] = pv;
            #pragma unroll
            for (int msk = 1; msk <= 8; msk <<= 1)
                lsum += __shfl_xor(lsum, msk, 32);
            if (lm == 0) {
                rowM[row] = mnew;
                rowL[row] = rowL[row] * scl + lsum;
                rowScale[row] = scl;
            }
        }
        __syncthreads();

        // ---- rescale accumulators, then O += P(16x128) @ V(128x32) ----
        #pragma unroll
        for (int r = 0; r < 8; ++r) {
            float s = rowScale[r + 8 * hh];
            acc0[r] *= s; acc1[r] *= s;
        }
        const _Float16* v0p = v0Base + ch * 128;
        const _Float16* v1p = v1Base + ch * 128;
        #pragma unroll
        for (int kk = 0; kk < 4; ++kk) {
            int kb = kk * 32 + hh * 8;
            v16h A  = combine16(*(const v8h*)&sP[lm][kb],
                                *(const v8h*)&sP[lm][kb + 16]);
            v16h B0 = combine16(*(const v8h*)&v0p[kb], *(const v8h*)&v0p[kb + 16]);
            v16h B1 = combine16(*(const v8h*)&v1p[kb], *(const v8h*)&v1p[kb + 16]);
            acc0 = wmma_f16(A, B0, acc0);
            acc1 = wmma_f16(A, B1, acc1);
        }
        // next chunk's sS/sP writes are fenced by the two barriers above.
    }

    // ---- finalize: O = acc / l, fold rows (mean-pool partial), atomicAdd ----
    float cs0 = 0.f, cs1 = 0.f;
    #pragma unroll
    for (int r = 0; r < 8; ++r) {
        float inv = 1.f / rowL[r + 8 * hh];
        cs0 += acc0[r] * inv;
        cs1 += acc1[r] * inv;
    }
    cs0 += __shfl_xor(cs0, 16, 32);   // combine row groups 0-7 and 8-15
    cs1 += __shfl_xor(cs1, 16, 32);
    if (lane < 16) {
        atomicAdd(&pooled[b * EMBD + hcol + lm], cs0);
        atomicAdd(&pooled[b * EMBD + hcol + 16 + lm], cs1);
    }
}

// ---------------------------------------------------------------------------
// Kernel 3: out[b,c] = mean_s(out)[b,:] @ Wfc + bfc   (32x2, trivial)
// ---------------------------------------------------------------------------
__global__ void fc_kernel(const float* __restrict__ pooled,
                          const float* __restrict__ Wfc,
                          const float* __restrict__ bfc,
                          float* __restrict__ out)
{
    int t = threadIdx.x;
    if (t >= 64) return;
    int bb = t >> 1, c = t & 1;
    float s = 0.f;
    for (int h = 0; h < EMBD; ++h)
        s += pooled[bb * EMBD + h] * Wfc[h * 2 + c];
    out[bb * 2 + c] = s * (1.f / (float)SEQ) + bfc[c];
}

extern "C" void kernel_launch(void* const* d_in, const int* in_sizes, int n_in,
                              void* d_out, int out_size, void* d_ws, size_t ws_size,
                              hipStream_t stream) {
    const int*   x   = (const int*)d_in[0];
    const float* emb = (const float*)d_in[1];
    const float* Wq  = (const float*)d_in[2];
    const float* bq  = (const float*)d_in[3];
    const float* Wk  = (const float*)d_in[4];
    const float* bk  = (const float*)d_in[5];
    const float* Wv  = (const float*)d_in[6];
    const float* bv  = (const float*)d_in[7];
    const float* Wfc = (const float*)d_in[8];
    const float* bfc = (const float*)d_in[9];
    float* out = (float*)d_out;

    const size_t QKV = (size_t)32 * SEQ * EMBD;   // 16,777,216 elements each
    _Float16* Qh  = (_Float16*)d_ws;
    _Float16* Kh  = Qh + QKV;
    _Float16* Vt  = Kh + QKV;                     // [b][h][s] transposed
    _Float16* WtQ = Vt + QKV;
    _Float16* WtK = WtQ + 65536;
    _Float16* WtV = WtK + 65536;
    float* pooled = (float*)(WtV + 65536);

    hipMemsetAsync(pooled, 0, 32 * EMBD * sizeof(float), stream);
    prep_kernel<<<dim3(768), dim3(256), 0, stream>>>(Wq, Wk, Wv, WtQ, WtK, WtV);
    qkv_kernel<<<dim3((32 * SEQ) / 16), dim3(256), 0, stream>>>(
        x, emb, WtQ, WtK, WtV, bq, bk, bv, Qh, Kh, Vt);
    attn_kernel<<<dim3(SEQ / 16, 32), dim3(256), 0, stream>>>(Qh, Kh, Vt, pooled);
    fc_kernel<<<dim3(1), dim3(64), 0, stream>>>(pooled, Wfc, bfc, out);
}
